// Criterion_24489903522255
// MI455X (gfx1250) — matrix operand order repaired and checked by reading, not compiled, not run on hardware
//
#include <hip/hip_runtime.h>
#include <hip/hip_bf16.h>

// ---------------------------------------------------------------------------
// Problem: BS=4096, C=16384, D=128.  Fused cosine-sim GEMM + column-wise
// masked online-logsumexp + softplus + masked mean -> scalar.
// Precision: inputs normalized to unit vectors -> f16 fragments feed
// v_wmma_f32_16x16x32_f16 (f32 accumulate).
// Row loop is software-pipelined (double-buffered A fragments) so L2 load
// latency overlaps the WMMA chain.
// ---------------------------------------------------------------------------

typedef _Float16 v16h __attribute__((ext_vector_type(16)));
typedef _Float16 v8h  __attribute__((ext_vector_type(8)));
typedef _Float16 v4h  __attribute__((ext_vector_type(4)));
typedef float    v8f  __attribute__((ext_vector_type(8)));

#define BS_N 4096
#define C_N  16384
#define D_N  128

__device__ __forceinline__ v16h load_frag16(const _Float16* p0, const _Float16* p1) {
    v8h a = *(const v8h*)p0;   // 16B contiguous
    v8h b = *(const v8h*)p1;   // 16B contiguous
    v16h r;
#pragma unroll
    for (int i = 0; i < 8; ++i) { r[i] = a[i]; r[i + 8] = b[i]; }
    return r;
}

// A tile fragments for one 16x16x128 stack: lane's row = arow, K-segments
// [kk*32 + hi*8, +8) and [kk*32 + 16 + hi*8, +8).
__device__ __forceinline__ void load_atile(const _Float16* arow, int hi, v16h frag[4]) {
#pragma unroll
    for (int kk = 0; kk < 4; ++kk) {
        const _Float16* p = arow + kk * 32 + hi * 8;
        frag[kk] = load_frag16(p, p + 16);
    }
}

__device__ __forceinline__ v8f do_wmma4(const v16h a[4], const v16h b[4]) {
    v8f c = {};
#pragma unroll
    for (int kk = 0; kk < 4; ++kk)
        c = __builtin_amdgcn_wmma_f32_16x16x32_f16(
                false, a[kk], false, b[kk], (short)0, c, false, false);
    return c;
}

// Online masked LSE update for one 16x16 tile held in accumulator c.
template <bool IS_POS>
__device__ __forceinline__ void tile_epilogue(const v8f& c, int4 la, int4 lb,
                                              int col, int collab, float WSC,
                                              float& M, float& mmin, float& s) {
    int rl[8] = {la.x, la.y, la.z, la.w, lb.x, lb.y, lb.z, lb.w};
    float w[8], m[8];
    bool  in[8];
#pragma unroll
    for (int r = 0; r < 8; ++r) {
        w[r]  = fmaf(c[r], WSC, 3.2f);
        in[r] = IS_POS ? (rl[r] == collab) : (rl[r] != col);
        m[r]  = in[r] ? w[r] : 0.0f;
    }
    float tmax = fmaxf(fmaxf(fmaxf(m[0], m[1]), fmaxf(m[2], m[3])),
                       fmaxf(fmaxf(m[4], m[5]), fmaxf(m[6], m[7])));
    float tmin = fminf(fminf(fminf(m[0], m[1]), fminf(m[2], m[3])),
                       fminf(fminf(m[4], m[5]), fminf(m[6], m[7])));
    mmin = fminf(mmin, tmin);
    float newM = fmaxf(M, tmax);
    s *= __expf(M - newM);          // one rescale per tile; M=-inf start: s=0
    M = newM;
    float e[8];
#pragma unroll
    for (int r = 0; r < 8; ++r)     // 8 independent TRANS ops
        e[r] = in[r] ? __expf(w[r] - M) : 0.0f;
    s += ((e[0] + e[1]) + (e[2] + e[3])) + ((e[4] + e[5]) + (e[6] + e[7]));
}

// --- row L2-normalize + f32 -> f16 convert; one wave per 128-wide row -------
__global__ void normalize_rows_kernel(const float* __restrict__ src,
                                      _Float16* __restrict__ dst, int nrows) {
    int gtid = blockIdx.x * blockDim.x + threadIdx.x;
    int wid  = gtid >> 5;
    int lane = gtid & 31;
    if (wid >= nrows) return;
    float4 v = ((const float4*)(src + (size_t)wid * D_N))[lane];
    float ss = v.x * v.x + v.y * v.y + v.z * v.z + v.w * v.w;
#pragma unroll
    for (int off = 16; off >= 1; off >>= 1) ss += __shfl_xor(ss, off, 32);
    float inv = rsqrtf(ss);
    v4h h;
    h[0] = (_Float16)(v.x * inv); h[1] = (_Float16)(v.y * inv);
    h[2] = (_Float16)(v.z * inv); h[3] = (_Float16)(v.w * inv);
    *(v4h*)(dst + (size_t)wid * D_N + lane * 4) = h;   // 8B aligned store
}

__global__ void zero_acc_kernel(float* acc) {
    if (threadIdx.x < 4 && blockIdx.x == 0) acc[threadIdx.x] = 0.0f;
}

// --- fused GEMM + column-wise online masked LSE + softplus ------------------
// One wave owns 16 columns, sweeps all 4096 rows in 16x16 tiles (x2 unrolled,
// double-buffered).  IS_POS: columns are anchors j -> B row = labels[j],
// mask-in rowlab==collab, w = -32*sim + 3.2.  !IS_POS: columns are classes j
// -> B row = j, mask-in rowlab!=j, w = 32*sim + 3.2.
template <bool IS_POS>
__global__ void gemm_lse_kernel(const _Float16* __restrict__ A,   // [4096,128]
                                const _Float16* __restrict__ P,   // [16384,128]
                                const int* __restrict__ labels,   // [4096]
                                float* __restrict__ acc,          // [4]
                                int ncols) {
    const int lane = threadIdx.x & 31;
    const int wid  = (blockIdx.x * blockDim.x + threadIdx.x) >> 5;
    const int colbase = wid * 16;
    if (colbase >= ncols) return;               // wave-uniform: EXEC stays full
    const int nlo = lane & 15;
    const int hi  = lane >> 4;                  // 0 or 1
    const int col = colbase + nlo;              // this lane's column

    int collab = 0;
    int brow;
    if (IS_POS) { collab = labels[col]; brow = collab; }
    else        { brow = col; }

    // B fragments: proxy row 'brow', K-segments [kk*32 + hi*8, +8) and [kk*32 + 16 + hi*8, +8), held in regs.
    const _Float16* prow = P + (size_t)brow * D_N;
    v16h bfrag[4];
#pragma unroll
    for (int kk = 0; kk < 4; ++kk) {
        const _Float16* p = prow + kk * 32 + hi * 8;
        bfrag[kk] = load_frag16(p, p + 16);
    }

    const float WSC = IS_POS ? -32.0f : 32.0f;
    float M = -INFINITY;   // running max of m = w*mask (masked-out -> 0)
    float mmin = INFINITY; // running min of m (for the nz filter)
    float s = 0.0f;        // running sum of exp(w - M) over mask-in entries

    const _Float16* arow0 = A + (size_t)nlo * D_N;   // lane's row in tile 0
    v16h bufA[4], bufB[4];
    load_atile(arow0, hi, bufA);                     // prologue: tile 0

    for (int t = 0; t < BS_N; t += 32) {
        // Row labels (contiguous per half-lane group; L0-broadcast loads).
        const int4* lp0 = (const int4*)(labels + t + hi * 8);
        int4 la0 = lp0[0], lb0 = lp0[1];
        const int4* lp1 = (const int4*)(labels + t + 16 + hi * 8);
        int4 la1 = lp1[0], lb1 = lp1[1];

        __builtin_prefetch(arow0 + (size_t)(t + 48) * D_N, 0, 3);

        // Issue loads for tile t+16, then consume tile t.
        load_atile(arow0 + (size_t)(t + 16) * D_N, hi, bufB);
        v8f c0 = do_wmma4(bufA, bfrag);
        tile_epilogue<IS_POS>(c0, la0, lb0, col, collab, WSC, M, mmin, s);

        // Issue loads for tile t+32 (last iter reads 16 rows past A: still
        // inside the workspace allocation, values unused), consume tile t+16.
        load_atile(arow0 + (size_t)(t + 32) * D_N, hi, bufA);
        v8f c1 = do_wmma4(bufB, bfrag);
        tile_epilogue<IS_POS>(c1, la1, lb1, col, collab, WSC, M, mmin, s);
    }

    // Merge the two half-lanes that share a column (xor 16).
    {
        float oM = __shfl_xor(M, 16, 32);
        float os = __shfl_xor(s, 16, 32);
        float om = __shfl_xor(mmin, 16, 32);
        float nM = fmaxf(M, oM);
        float e0 = (M == nM) ? 1.0f : __expf(M - nM);
        float e1 = (oM == nM) ? 1.0f : __expf(oM - nM);
        s = s * e0 + os * e1;
        M = nM;
        mmin = fminf(mmin, om);
    }

    // Per-column softplus(lse), filtered by nz = (max+min)!=0.
    float val = 0.0f, cnt = 0.0f;
    if ((M + mmin) != 0.0f) {
        float lse = __logf(s) + M;
        val = (lse > 20.0f) ? lse : log1pf(__expf(lse));
        cnt = 1.0f;
    }
    // Butterfly over the 16 columns (lanes l and l+16 are duplicates).
#pragma unroll
    for (int off = 1; off <= 8; off <<= 1) {
        val += __shfl_xor(val, off, 32);
        cnt += __shfl_xor(cnt, off, 32);
    }
    if (lane == 0) {
        atomicAdd(acc + (IS_POS ? 0 : 2), val);
        atomicAdd(acc + (IS_POS ? 1 : 3), cnt);
    }
}

__global__ void finalize_kernel(const float* __restrict__ acc,
                                float* __restrict__ out) {
    if (threadIdx.x == 0 && blockIdx.x == 0) {
        float pm = acc[0] / fmaxf(acc[1], 1.0f);
        float nm = acc[2] / fmaxf(acc[3], 1.0f);
        out[0] = pm + nm;
    }
}

extern "C" void kernel_launch(void* const* d_in, const int* in_sizes, int n_in,
                              void* d_out, int out_size, void* d_ws, size_t ws_size,
                              hipStream_t stream) {
    (void)in_sizes; (void)n_in; (void)out_size; (void)ws_size;
    const float* batch   = (const float*)d_in[0];   // [4096,128] f32
    const float* proxies = (const float*)d_in[1];   // [16384,128] f32
    const int*   labels  = (const int*)d_in[2];     // [4096] i32
    float* out = (float*)d_out;

    // Workspace layout: [0..16) acc floats; then f16 matrices (256B aligned).
    // batchN is immediately followed by proxN, so the pipelined one-tile
    // overrun read past batchN stays inside the allocation.
    char* ws = (char*)d_ws;
    float*    acc    = (float*)ws;
    _Float16* batchN = (_Float16*)(ws + 256);
    _Float16* proxN  = (_Float16*)(ws + 256 + (size_t)BS_N * D_N * sizeof(_Float16));

    zero_acc_kernel<<<1, 32, 0, stream>>>(acc);

    {   // normalize + convert: one wave per row
        int threads = 256;
        int blkB = (BS_N * 32 + threads - 1) / threads;
        int blkP = (C_N  * 32 + threads - 1) / threads;
        normalize_rows_kernel<<<blkB, threads, 0, stream>>>(batch,   batchN, BS_N);
        normalize_rows_kernel<<<blkP, threads, 0, stream>>>(proxies, proxN,  C_N);
    }

    {   // neg: 16384 columns -> 1024 waves; pos: 4096 columns -> 256 waves
        int threads = 256;                       // 8 waves / block
        int blkNeg = (C_N  / 16) * 32 / threads; // 128 blocks
        int blkPos = (BS_N / 16) * 32 / threads; // 32 blocks
        gemm_lse_kernel<false><<<blkNeg, threads, 0, stream>>>(batchN, proxN, labels, acc, C_N);
        gemm_lse_kernel<true ><<<blkPos, threads, 0, stream>>>(batchN, proxN, labels, acc, BS_N);
    }

    finalize_kernel<<<1, 32, 0, stream>>>(acc, out);
}
